// AtlasMemory_21182778704935
// MI455X (gfx1250) — matrix operand order repaired and checked by reading, not compiled, not run on hardware
//
#include <hip/hip_runtime.h>
#include <hip/hip_bf16.h>
#include <math.h>

#define D 256
#define W 8
#define LSEQ 256
#define BATCH 4
#define NS_EPS 1e-7f

typedef __bf16 bf16;
typedef __attribute__((ext_vector_type(16))) __bf16 v16bf;
typedef __attribute__((ext_vector_type(8)))  __bf16 v8bf;
typedef __attribute__((ext_vector_type(8)))  float  v8f;

__device__ __forceinline__ float sigmoidf_(float z) { return 1.0f / (1.0f + __expf(-z)); }

// ---------------------------------------------------------------------------
// Kernel 1: token-parallel precompute of gates (alpha/eta/theta/gamma) and
// polynomial features phi(k). One block per token, 256 threads (thread = d).
// ---------------------------------------------------------------------------
__global__ void gates_kernel(const float* __restrict__ x,
                             const float* __restrict__ k_aligned,
                             const float* __restrict__ poly,      // [2][D]
                             const float* __restrict__ aW, const float* __restrict__ ab,
                             const float* __restrict__ eW, const float* __restrict__ eb,
                             const float* __restrict__ tW, const float* __restrict__ tb,
                             const float* __restrict__ gW, const float* __restrict__ gb,
                             float* __restrict__ alpha, float* __restrict__ eta,
                             float* __restrict__ theta, float* __restrict__ gamma,
                             float* __restrict__ kphi)
{
    __shared__ float xs[D];
    __shared__ float red[D];
    const int t = blockIdx.x;       // flat token 0..B*L-1
    const int d = threadIdx.x;
    const float xv = x[(size_t)t * D + d];
    xs[d] = xv;
    __syncthreads();

    const float4* aRow = (const float4*)(aW + (size_t)d * D);
    const float4* eRow = (const float4*)(eW + (size_t)d * D);
    const float4* tRow = (const float4*)(tW + (size_t)d * D);
    const float4* x4   = (const float4*)xs;
    float sa = 0.f, se = 0.f, st = 0.f;
#pragma unroll 4
    for (int e4 = 0; e4 < D / 4; ++e4) {
        const float4 xe = x4[e4];
        const float4 av = aRow[e4], ev = eRow[e4], tv = tRow[e4];
        sa += xe.x * av.x + xe.y * av.y + xe.z * av.z + xe.w * av.w;
        se += xe.x * ev.x + xe.y * ev.y + xe.z * ev.z + xe.w * ev.w;
        st += xe.x * tv.x + xe.y * tv.y + xe.z * tv.z + xe.w * tv.w;
    }
    alpha[(size_t)t * D + d] = sigmoidf_(sa + ab[d]);
    eta[(size_t)t * D + d]   = 0.1f * sigmoidf_(se + eb[d]);
    theta[(size_t)t * D + d] = sigmoidf_(st + tb[d]);

    const float kv = k_aligned[(size_t)t * D + d];
    kphi[(size_t)t * D + d] = poly[d] * kv + poly[D + d] * kv * kv;

    // gamma[t] = sigmoid(<x, gW> + gb)
    red[d] = xv * gW[d];
    __syncthreads();
    for (int s = D / 2; s > 0; s >>= 1) {
        if (d < s) red[d] += red[d + s];
        __syncthreads();
    }
    if (d == 0) gamma[t] = sigmoidf_(red[0] + gb[0]);
}

// ---------------------------------------------------------------------------
// Kernel 2: the sequential scan. One workgroup (1024 threads = 32 waves) per
// batch. M,S state lives in global (the d_out M/S regions, L2-resident).
// X = S/||S||_F and X^T live in LDS as bf16 and feed v_wmma_f32_16x16x32_bf16.
// All cross-wave global RAWs are same-WGP -> workgroup-scope fences only.
// ---------------------------------------------------------------------------
__global__ void __launch_bounds__(1024, 1)
scan_kernel(const float* __restrict__ Mprev, const float* __restrict__ Sprev,
            const float* __restrict__ kphi_g, const float* __restrict__ v_g,
            const float* __restrict__ alpha_g, const float* __restrict__ eta_g,
            const float* __restrict__ theta_g, const float* __restrict__ gamma_g,
            float* __restrict__ M_g, float* __restrict__ S_g,   // [B][D][D] live state
            float* __restrict__ ys_g,                           // [B*L][D]
            bf16* __restrict__ T_g)                             // [B][D][D] bf16 scratch
{
    __shared__ __align__(16) bf16 Xs[D][D];    // 128 KB: X (row-major)
    __shared__ __align__(16) bf16 XTs[D][D];   // 128 KB: X^T (row-major)
    __shared__ __align__(16) float bk[W][D];   // sliding window of phi(k)
    __shared__ __align__(16) float bv[W][D];   // sliding window of v
    __shared__ __align__(16) float werr[W][D]; // g_w * err[w][d] / c
    __shared__ float bg[W];
    __shared__ float alpha_s[D], eta_s[D], theta_s[D];
    __shared__ float y_s[D];
    __shared__ float red[32];
    __shared__ float rinv_s;

    const int b    = blockIdx.x;
    const int tid  = threadIdx.x;
    const int lane = tid & 31;
    const int wv   = tid >> 5;        // wave 0..31
    const int hi   = lane >> 4;       // half-wave 0/1
    const int l15  = lane & 15;

    float* Mb = M_g + (size_t)b * D * D;
    float* Sb = S_g + (size_t)b * D * D;
    bf16*  Tb = T_g + (size_t)b * D * D;

    // ---- init state from M_prev / S_prev; zero window + y accumulator ----
    for (int i = tid; i < D * D; i += 1024) {
        Mb[i] = Mprev[(size_t)b * D * D + i];
        Sb[i] = Sprev[(size_t)b * D * D + i];
    }
    for (int i = tid; i < W * D; i += 1024) { bk[i >> 8][i & 255] = 0.f; bv[i >> 8][i & 255] = 0.f; }
    if (tid < W) bg[tid] = 0.f;
    if (tid < D) y_s[tid] = 0.f;
    __threadfence_block();
    __syncthreads();

    // tile assignment: wave wv covers 8 tiles in row-block m0, cols n0base+j*16
    const int m0     = (wv >> 1) << 4;
    const int n0base = (wv & 1) * 128;

    for (int t = 0; t < LSEQ; ++t) {
        const int   tok  = b * LSEQ + t;
        const int   slot = t & (W - 1);
        const float c_t  = (float)((t + 1 < W) ? (t + 1) : W);

        // ---- stage token data; circular window insert ----
        if (tid < D) {
            bk[slot][tid]  = kphi_g[(size_t)tok * D + tid];
            bv[slot][tid]  = v_g[(size_t)tok * D + tid];
            alpha_s[tid]   = alpha_g[(size_t)tok * D + tid];
            eta_s[tid]     = eta_g[(size_t)tok * D + tid];
            theta_s[tid]   = theta_g[(size_t)tok * D + tid];
        }
        if (tid == 0) bg[slot] = gamma_g[tok];
        // prefetch next token's rows (one lane per 128B cacheline)
        if (t + 1 < LSEQ && tid < D && (tid & 31) == 0) {
            const size_t nt = (size_t)(tok + 1) * D + tid;
            __builtin_prefetch(&kphi_g[nt], 0, 1);
            __builtin_prefetch(&v_g[nt], 0, 1);
            __builtin_prefetch(&alpha_g[nt], 0, 1);
            __builtin_prefetch(&eta_g[nt], 0, 1);
            __builtin_prefetch(&theta_g[nt], 0, 1);
        }
        __syncthreads();

        // ---- pred/err over window: werr[w][d] = g_w*(M[d,:]·bk[w,:] - bv[w][d])/c ----
#pragma unroll
        for (int rep = 0; rep < 2; ++rep) {
            const int idx = tid + rep * 1024;       // 2048 (w,d) entries
            const int w   = idx >> 8;
            const int dd  = idx & 255;
            const float4* m4 = (const float4*)(Mb + (size_t)dd * D);
            const float4* b4 = (const float4*)&bk[w][0];
            float acc = 0.f;
#pragma unroll 4
            for (int e4 = 0; e4 < D / 4; ++e4) {
                const float4 mv = m4[e4];
                const float4 kvv = b4[e4];
                acc += mv.x * kvv.x + mv.y * kvv.y + mv.z * kvv.z + mv.w * kvv.w;
            }
            werr[w][dd] = bg[w] * (acc - bv[w][dd]) / c_t;
        }
        __syncthreads();

        // ---- S <- theta[row]*S + grad (rank-8); Frobenius norm; keep in regs ----
        const int srow = tid >> 2;
        const int scol = (tid & 3) * 64;
        float sreg[64];                 // grad accumulator, then updated S values
        float ssum = 0.f;
        {
#pragma unroll
            for (int j = 0; j < 64; ++j) sreg[j] = 0.f;
#pragma unroll
            for (int w = 0; w < W; ++w) {           // w-outer: vectorized bk reads
                const float wrw = werr[w][srow];
                const float4* bk4 = (const float4*)&bk[w][scol];
#pragma unroll
                for (int q = 0; q < 16; ++q) {
                    const float4 bb = bk4[q];
                    sreg[4*q+0] += wrw * bb.x;
                    sreg[4*q+1] += wrw * bb.y;
                    sreg[4*q+2] += wrw * bb.z;
                    sreg[4*q+3] += wrw * bb.w;
                }
            }
            const float th = theta_s[srow];
            float4* sptr4 = (float4*)(Sb + (size_t)srow * D + scol);
#pragma unroll
            for (int q = 0; q < 16; ++q) {
                float4 sv = sptr4[q];
                sv.x = th * sv.x + sreg[4*q+0];
                sv.y = th * sv.y + sreg[4*q+1];
                sv.z = th * sv.z + sreg[4*q+2];
                sv.w = th * sv.w + sreg[4*q+3];
                sptr4[q] = sv;
                sreg[4*q+0] = sv.x; sreg[4*q+1] = sv.y;
                sreg[4*q+2] = sv.z; sreg[4*q+3] = sv.w;
                ssum += sv.x*sv.x + sv.y*sv.y + sv.z*sv.z + sv.w*sv.w;
            }
        }
#pragma unroll
        for (int off = 16; off > 0; off >>= 1) ssum += __shfl_xor(ssum, off, 32);
        if (lane == 0) red[wv] = ssum;
        __syncthreads();
        if (tid == 0) {
            float tot = 0.f;
#pragma unroll
            for (int i = 0; i < 32; ++i) tot += red[i];
            rinv_s = 1.0f / (sqrtf(tot) + NS_EPS);
        }
        __syncthreads();
        const float rinv = rinv_s;

        // ---- X = S * rinv -> LDS (bf16): packed b128 row stores + scalar X^T scatter ----
#pragma unroll
        for (int q = 0; q < 8; ++q) {
            v8bf pack;
#pragma unroll
            for (int u = 0; u < 8; ++u) pack[u] = (bf16)(sreg[8*q+u] * rinv);
            *(v8bf*)&Xs[srow][scol + 8*q] = pack;
#pragma unroll
            for (int u = 0; u < 8; ++u) XTs[scol + 8*q + u][srow] = pack[u];
        }
        __syncthreads();

        // ---- GEMM1: T = X * X^T  (bf16 WMMA, f32 acc), T -> global bf16 ----
        {
            v8f acc[8] = {};
            for (int k0 = 0; k0 < D; k0 += 32) {
                union { v16bf v; v8bf h[2]; } A;
                A.h[0] = *(const v8bf*)&Xs[m0 + l15][k0 + hi * 8];
                A.h[1] = *(const v8bf*)&Xs[m0 + l15][k0 + 16 + hi * 8];
#pragma unroll
                for (int j = 0; j < 8; ++j) {
                    const int n = n0base + j * 16 + l15;
                    union { v16bf v; v8bf h[2]; } Bf;
                    Bf.h[0] = *(const v8bf*)&Xs[n][k0 + hi * 8];
                    Bf.h[1] = *(const v8bf*)&Xs[n][k0 + 16 + hi * 8];
                    acc[j] = __builtin_amdgcn_wmma_f32_16x16x32_bf16(
                        false, A.v, false, Bf.v, (short)0, acc[j], false, false);
                }
            }
#pragma unroll
            for (int j = 0; j < 8; ++j) {
                const int n = n0base + j * 16 + l15;
#pragma unroll
                for (int r = 0; r < 8; ++r) {
                    const int row = m0 + r + 8 * hi;
                    Tb[(size_t)row * D + n] = (bf16)acc[j][r];
                }
            }
        }
        __threadfence_block();      // same-WGP visibility of T is enough
        __syncthreads();

        // ---- GEMM2: R = T * X; fuse S_ortho = 1.5X - 0.5R, M update, y = M·k ----
        {
            v8f acc[8] = {};
            for (int k0 = 0; k0 < D; k0 += 32) {
                union { v16bf v; v8bf h[2]; } A;
                A.h[0] = *(const v8bf*)&Tb[(size_t)(m0 + l15) * D + k0 + hi * 8];
                A.h[1] = *(const v8bf*)&Tb[(size_t)(m0 + l15) * D + k0 + 16 + hi * 8];
#pragma unroll
                for (int j = 0; j < 8; ++j) {
                    const int n = n0base + j * 16 + l15;
                    union { v16bf v; v8bf h[2]; } Bf;
                    Bf.h[0] = *(const v8bf*)&XTs[n][k0 + hi * 8];
                    Bf.h[1] = *(const v8bf*)&XTs[n][k0 + 16 + hi * 8];
                    acc[j] = __builtin_amdgcn_wmma_f32_16x16x32_bf16(
                        false, A.v, false, Bf.v, (short)0, acc[j], false, false);
                }
            }
#pragma unroll
            for (int j = 0; j < 8; ++j) {
                const int   n  = n0base + j * 16 + l15;
                const float kn = bk[slot][n];     // current phi(k)[n]
#pragma unroll
                for (int r = 0; r < 8; ++r) {
                    const int   row  = m0 + r + 8 * hi;
                    const float xmn  = (float)Xs[row][n];
                    const float so   = 1.5f * xmn - 0.5f * acc[j][r];
                    const float mold = Mb[(size_t)row * D + n];
                    const float mnew = alpha_s[row] * mold - eta_s[row] * so;
                    Mb[(size_t)row * D + n] = mnew;
                    float py = mnew * kn;
#pragma unroll
                    for (int off = 1; off < 16; off <<= 1) py += __shfl_xor(py, off, 32);
                    if (l15 == 0) atomicAdd(&y_s[row], py);
                }
            }
        }
        __threadfence_block();      // make new M visible for next step's pred phase
        __syncthreads();
        if (tid < D) { ys_g[(size_t)tok * D + tid] = y_s[tid]; y_s[tid] = 0.f; }
        __syncthreads();
    }
}

// ---------------------------------------------------------------------------
// Kernel 3: output projection, one block per token.
// ---------------------------------------------------------------------------
__global__ void outproj_kernel(const float* __restrict__ ys,
                               const float* __restrict__ oW, const float* __restrict__ ob,
                               float* __restrict__ out)
{
    __shared__ float xs[D];
    const int t = blockIdx.x;
    const int d = threadIdx.x;
    xs[d] = ys[(size_t)t * D + d];
    __syncthreads();
    const float4* row = (const float4*)(oW + (size_t)d * D);
    const float4* x4  = (const float4*)xs;
    float acc = 0.f;
#pragma unroll 4
    for (int e4 = 0; e4 < D / 4; ++e4) {
        const float4 xv = x4[e4];
        const float4 wv = row[e4];
        acc += xv.x * wv.x + xv.y * wv.y + xv.z * wv.z + xv.w * wv.w;
    }
    out[(size_t)t * D + d] = acc + ob[d];
}

// ---------------------------------------------------------------------------
extern "C" void kernel_launch(void* const* d_in, const int* in_sizes, int n_in,
                              void* d_out, int out_size, void* d_ws, size_t ws_size,
                              hipStream_t stream)
{
    const float* x     = (const float*)d_in[0];
    const float* k_al  = (const float*)d_in[1];
    const float* v     = (const float*)d_in[2];
    const float* Mprev = (const float*)d_in[3];
    const float* Sprev = (const float*)d_in[4];
    const float* poly  = (const float*)d_in[5];
    const float* aW = (const float*)d_in[6];  const float* ab = (const float*)d_in[7];
    const float* eW = (const float*)d_in[8];  const float* eb = (const float*)d_in[9];
    const float* tW = (const float*)d_in[10]; const float* tb = (const float*)d_in[11];
    const float* gW = (const float*)d_in[12]; const float* gb = (const float*)d_in[13];
    const float* oW = (const float*)d_in[14]; const float* ob = (const float*)d_in[15];

    float* out  = (float*)d_out;                      // [B,L,D]
    float* Mout = out + (size_t)BATCH * LSEQ * D;     // [B,D,D] final M (live state)
    float* Sout = Mout + (size_t)BATCH * D * D;       // [B,D,D] final S (live state)

    float* ws    = (float*)d_ws;
    float* kphi  = ws;                                // B*L*D
    float* alpha = kphi  + (size_t)BATCH * LSEQ * D;  // B*L*D
    float* eta   = alpha + (size_t)BATCH * LSEQ * D;  // B*L*D
    float* theta = eta   + (size_t)BATCH * LSEQ * D;  // B*L*D
    float* gamma = theta + (size_t)BATCH * LSEQ * D;  // B*L
    float* ysbuf = gamma + (size_t)BATCH * LSEQ;      // B*L*D
    bf16*  Tbuf  = (bf16*)(ysbuf + (size_t)BATCH * LSEQ * D); // B*D*D bf16

    gates_kernel<<<BATCH * LSEQ, D, 0, stream>>>(x, k_al, poly, aW, ab, eW, eb, tW, tb,
                                                 gW, gb, alpha, eta, theta, gamma, kphi);
    scan_kernel<<<BATCH, 1024, 0, stream>>>(Mprev, Sprev, kphi, v, alpha, eta, theta, gamma,
                                            Mout, Sout, ysbuf, Tbuf);
    outproj_kernel<<<BATCH * LSEQ, D, 0, stream>>>(ysbuf, oW, ob, out);
}
